// PCA_LSTM_80290118632273
// MI455X (gfx1250) — matrix operand order repaired and checked by reading, not compiled, not run on hardware
//
#include <hip/hip_runtime.h>

// ---------------------------------------------------------------------------
// PCA + LSTM encoder/decoder for MI455X (gfx1250, wave32, WMMA).
// bf16 WMMA (16x16x32, f32 accum) everywhere; weights converted once to bf16
// (64MB, L2-resident). Each recurrent step = ONE fused kernel:
//   z = [x;h] @ Wcat (K=4096, K split across waves) -> LDS -> gates -> c,h.
// Inner GEMM loops are software-pipelined with 2 independent wmma chains per
// wave so next-tile loads fly under the current wmmas.
// ---------------------------------------------------------------------------

typedef __attribute__((ext_vector_type(16))) __bf16 v16bf;
typedef __attribute__((ext_vector_type(8)))  float  v8f;

#define N_UNITS 2048
#define FOUR_U  8192
#define KTOT    4096
#define BATCH   32
#define HSTEPS  64
#define PSTEPS  24

union FragBF {
    v16bf v;
    uint4 q[2];
};

// A fragment (16x32 bf16), row-major source, CDNA5 layout:
// lane = M (mod 16); lanes 0-15: K = {0..7,16..23}; lanes 16-31: K = {8..15,24..31}.
__device__ __forceinline__ FragBF load_a_frag(const __bf16* row_base, int lda, int lane) {
    int m  = lane & 15;
    int kb = (lane < 16) ? 0 : 8;
    const __bf16* p = row_base + (size_t)m * lda + kb;
    FragBF f;
    f.q[0] = *(const uint4*)(p);
    f.q[1] = *(const uint4*)(p + 16);
    return f;
}

// B fragment (32x16 bf16), row-major source: lane = N (mod 16); lanes 0-15: K = {0..7,16..23}; lanes 16-31: K = {8..15,24..31}.
__device__ __forceinline__ FragBF load_b_frag(const __bf16* p0, int ldb, int lane) {
    int n  = lane & 15;
    int kb = (lane < 16) ? 0 : 8;
    const __bf16* p = p0 + n;
    FragBF f;
#pragma unroll
    for (int s = 0; s < 8; ++s) {
        f.v[s]     = p[(size_t)(kb + s) * ldb];
        f.v[s + 8] = p[(size_t)(16 + kb + s) * ldb];
    }
    return f;
}

__device__ __forceinline__ v8f wmma_bf16(FragBF a, FragBF b, v8f c) {
    return __builtin_amdgcn_wmma_f32_16x16x32_bf16(false, a.v, false, b.v,
                                                   (short)0, c, false, false);
}

// ---------------- one-time prep kernels ----------------

__global__ void prep_xc_kernel(const float* __restrict__ in, const float* __restrict__ mean,
                               unsigned short* __restrict__ xc_raw) {
    __bf16* xc = (__bf16*)xc_raw;
    const int total = 2048 * 2048;
    for (int idx = blockIdx.x * blockDim.x + threadIdx.x; idx < total;
         idx += gridDim.x * blockDim.x) {
        int r = idx >> 11;          // t*32 + b
        int k = idx & 2047;
        int t = r >> 5;
        int b = r & 31;
        xc[idx] = (__bf16)(in[((size_t)b * HSTEPS + t) * N_UNITS + k] - mean[k]);
    }
}

__global__ void transpose_c_kernel(const float* __restrict__ comps,
                                   unsigned short* __restrict__ cbt_raw) {
    __bf16* cbt = (__bf16*)cbt_raw;
    const int total = 2048 * 2048;
    for (int idx = blockIdx.x * blockDim.x + threadIdx.x; idx < total;
         idx += gridDim.x * blockDim.x) {
        int k = idx >> 11;
        int n = idx & 2047;
        cbt[idx] = (__bf16)comps[(size_t)n * 2048 + k];
    }
}

__global__ void convert_w_kernel(const float* __restrict__ kern, const float* __restrict__ rec,
                                 unsigned short* __restrict__ w_raw) {
    __bf16* W = (__bf16*)w_raw;
    const size_t total = (size_t)KTOT * FOUR_U;
    for (size_t idx = (size_t)blockIdx.x * blockDim.x + threadIdx.x; idx < total;
         idx += (size_t)gridDim.x * blockDim.x) {
        size_t k = idx >> 13;
        size_t n = idx & 8191;
        float v = (k < N_UNITS) ? kern[k * FOUR_U + n] : rec[(k - N_UNITS) * FOUR_U + n];
        W[idx] = (__bf16)v;
    }
}

__global__ void init_state_kernel(unsigned short* __restrict__ h0,
                                  float* __restrict__ c_state) {
    const int total = BATCH * N_UNITS;
    for (int idx = blockIdx.x * blockDim.x + threadIdx.x; idx < total;
         idx += gridDim.x * blockDim.x) {
        h0[idx] = 0;
        c_state[idx] = 0.f;
    }
}

// ---------------- PCA GEMM: xs = xc @ C^T (2048x2048x2048, bf16 out) ----------------
// Block tile 32(M) x 128(N); wave = (mtile, npair): 1 A-frag shared by 2 N-tiles.
__global__ __launch_bounds__(256, 2)
void pca_gemm_kernel(const unsigned short* __restrict__ xc_raw,
                     const unsigned short* __restrict__ cbt_raw,
                     unsigned short* __restrict__ xs_raw) {
    const __bf16* A  = (const __bf16*)xc_raw;
    const __bf16* Bm = (const __bf16*)cbt_raw;
    __bf16*       xs = (__bf16*)xs_raw;

    int lane  = threadIdx.x & 31;
    int wv    = threadIdx.x >> 5;
    int mtile = wv & 1;
    int npair = wv >> 1;
    int r0 = blockIdx.x * 32 + mtile * 16;
    int n0 = blockIdx.y * 128 + npair * 32;

    v8f acc0 = {}, acc1 = {};
    FragBF fa  = load_a_frag(A + (size_t)r0 * 2048, 2048, lane);
    FragBF fb0 = load_b_frag(Bm + n0,      2048, lane);
    FragBF fb1 = load_b_frag(Bm + n0 + 16, 2048, lane);
    for (int k0 = 0; k0 < 2048; k0 += 32) {
        int kp = (k0 + 32 < 2048) ? (k0 + 32) : k0;   // clamped preload
        FragBF na  = load_a_frag(A + (size_t)r0 * 2048 + kp, 2048, lane);
        FragBF nb0 = load_b_frag(Bm + (size_t)kp * 2048 + n0,      2048, lane);
        FragBF nb1 = load_b_frag(Bm + (size_t)kp * 2048 + n0 + 16, 2048, lane);
        int kf = (k0 + 128 < 2048) ? (k0 + 128) : 2047;
        __builtin_prefetch((const void*)(Bm + (size_t)(kf + lane) * 2048 + n0), 0, 3);
        acc0 = wmma_bf16(fa, fb0, acc0);
        acc1 = wmma_bf16(fa, fb1, acc1);
        fa = na; fb0 = nb0; fb1 = nb1;
    }
#pragma unroll
    for (int r = 0; r < 8; ++r) {
        int row = r0 + ((lane < 16) ? r : r + 8);
        int col = lane & 15;
        xs[(size_t)row * 2048 + n0 + col]      = (__bf16)acc0[r];
        xs[(size_t)row * 2048 + n0 + 16 + col] = (__bf16)acc1[r];
    }
}

// ---------------- fused LSTM step ----------------
// Block owns 16 units. 8 waves = 4 gates x 2 K-halves (khalf0 -> x, khalf1 -> h),
// each wave does both M-tiles (2 independent wmma chains sharing one B-frag).
// Serial depth per step: 64 pipelined K-iterations.
__global__ __launch_bounds__(256, 2)
void lstm_step_kernel(const unsigned short* __restrict__ a0_raw,   // x source (32x2048 bf16)
                      const unsigned short* __restrict__ a1_raw,   // h_in     (32x2048 bf16)
                      const unsigned short* __restrict__ w_raw,    // Wcat (4096x8192 bf16)
                      const float* __restrict__ bias,
                      float* __restrict__ c_state,
                      unsigned short* __restrict__ h_out_raw,
                      float* __restrict__ out, int out_bstride) {
    __shared__ float zbuf[8][BATCH][16];   // [gate*2+khalf][batch][unit]

    const __bf16* A0 = (const __bf16*)a0_raw;
    const __bf16* A1 = (const __bf16*)a1_raw;
    const __bf16* W  = (const __bf16*)w_raw;
    __bf16* h_out    = (__bf16*)h_out_raw;

    int lane  = threadIdx.x & 31;
    int wv    = threadIdx.x >> 5;
    int khalf = wv & 1;
    int gate  = wv >> 1;                   // 0=i 1=f 2=g 3=o
    int nb = blockIdx.x * 16;
    int n0 = gate * N_UNITS + nb;

    const __bf16* A = khalf ? A1 : A0;               // this wave's K-half input
    const __bf16* Wh = W + (size_t)khalf * N_UNITS * FOUR_U;

    v8f acc0 = {}, acc1 = {};                        // batch 0-15 / 16-31
    FragBF fa0 = load_a_frag(A, 2048, lane);
    FragBF fa1 = load_a_frag(A + 16 * 2048, 2048, lane);
    FragBF fb  = load_b_frag(Wh + n0, FOUR_U, lane);
    for (int k0 = 0; k0 < 2048; k0 += 32) {
        int kp = (k0 + 32 < 2048) ? (k0 + 32) : k0;  // clamped preload
        FragBF na0 = load_a_frag(A + kp, 2048, lane);
        FragBF na1 = load_a_frag(A + 16 * 2048 + kp, 2048, lane);
        FragBF nb  = load_b_frag(Wh + (size_t)kp * FOUR_U + n0, FOUR_U, lane);
        int kf = (k0 + 256 < 2048) ? (k0 + 256) : 2047;
        __builtin_prefetch((const void*)(Wh + (size_t)(kf + lane) * FOUR_U + n0), 0, 3);
        acc0 = wmma_bf16(fa0, fb, acc0);
        acc1 = wmma_bf16(fa1, fb, acc1);
        fa0 = na0; fa1 = na1; fb = nb;
    }
#pragma unroll
    for (int r = 0; r < 8; ++r) {
        int row = (lane < 16) ? r : r + 8;
        int col = lane & 15;
        zbuf[wv][row][col]      = acc0[r];
        zbuf[wv][row + 16][col] = acc1[r];
    }
    __syncthreads();

    for (int e = threadIdx.x; e < BATCH * 16; e += blockDim.x) {
        int b  = e >> 4;
        int nl = e & 15;
        int n  = nb + nl;
        float zi = zbuf[0][b][nl] + zbuf[1][b][nl] + bias[0 * N_UNITS + n];
        float zf = zbuf[2][b][nl] + zbuf[3][b][nl] + bias[1 * N_UNITS + n];
        float zg = zbuf[4][b][nl] + zbuf[5][b][nl] + bias[2 * N_UNITS + n];
        float zo = zbuf[6][b][nl] + zbuf[7][b][nl] + bias[3 * N_UNITS + n];
        float ig = 1.f / (1.f + __expf(-zi));
        float fg = 1.f / (1.f + __expf(-zf));
        float gg = tanhf(zg);
        float og = 1.f / (1.f + __expf(-zo));
        int ci = b * N_UNITS + n;
        float cn = fg * c_state[ci] + ig * gg;
        c_state[ci] = cn;
        float hn = og * tanhf(cn);
        h_out[ci] = (__bf16)hn;
        if (out) out[(size_t)b * out_bstride + n] = hn;
    }
}

// ---------------- host launcher ----------------
extern "C" void kernel_launch(void* const* d_in, const int* in_sizes, int n_in,
                              void* d_out, int out_size, void* d_ws, size_t ws_size,
                              hipStream_t stream) {
    (void)in_sizes; (void)n_in; (void)out_size; (void)ws_size;
    const float* inputs = (const float*)d_in[0];
    const float* comps  = (const float*)d_in[1];
    const float* mean   = (const float*)d_in[2];
    const float* kern   = (const float*)d_in[3];
    const float* rec    = (const float*)d_in[4];
    const float* bias   = (const float*)d_in[5];
    float* out = (float*)d_out;

    // workspace partition (~94 MB incl. 1MB speculative-overread pad)
    char* ws = (char*)d_ws;
    size_t off = 0;
    unsigned short* xc  = (unsigned short*)(ws + off); off += (size_t)2048 * 2048 * 2;
    unsigned short* cbt = (unsigned short*)(ws + off); off += (size_t)2048 * 2048 * 2;
    unsigned short* xs  = (unsigned short*)(ws + off); off += (size_t)HSTEPS * BATCH * 2048 * 2;
    unsigned short* W   = (unsigned short*)(ws + off); off += (size_t)KTOT * FOUR_U * 2;
    unsigned short* h0  = (unsigned short*)(ws + off); off += (size_t)BATCH * 2048 * 2;
    unsigned short* h1  = (unsigned short*)(ws + off); off += (size_t)BATCH * 2048 * 2;
    float*          cst = (float*)(ws + off);          off += (size_t)BATCH * 2048 * 4;
    off += (size_t)1 << 20;  // pad for clamped speculative reads/prefetch

    prep_xc_kernel    <<<4096, 256, 0, stream>>>(inputs, mean, xc);
    transpose_c_kernel<<<4096, 256, 0, stream>>>(comps, cbt);
    convert_w_kernel  <<<8192, 256, 0, stream>>>(kern, rec, W);
    init_state_kernel <<<256, 256, 0, stream>>>(h0, cst);

    pca_gemm_kernel<<<dim3(64, 16), 256, 0, stream>>>(xc, cbt, xs);

    unsigned short* hin = h0;
    unsigned short* hout = h1;
    for (int t = 0; t < HSTEPS; ++t) {
        lstm_step_kernel<<<128, 256, 0, stream>>>(
            xs + (size_t)t * BATCH * 2048, hin, W, bias, cst, hout, nullptr, 0);
        unsigned short* tmp = hin; hin = hout; hout = tmp;
    }
    for (int p = 0; p < PSTEPS; ++p) {
        lstm_step_kernel<<<128, 256, 0, stream>>>(
            hin, hin, W, bias, cst, hout, out + (size_t)p * 2048, PSTEPS * 2048);
        unsigned short* tmp = hin; hin = hout; hout = tmp;
    }
}
